// QMHALayer_25735444038264
// MI455X (gfx1250) — matrix operand order repaired and checked by reading, not compiled, not run on hardware
//
#include <hip/hip_runtime.h>
#include <hip/hip_bf16.h>

#define D_MODEL 512
#define NUM_HEADS 8
#define D_HEAD 64
#define T_SEQ 4096
#define B_BATCH 2
#define MROWS (B_BATCH * T_SEQ)   // 8192

typedef __attribute__((ext_vector_type(16))) _Float16 v16h;
typedef __attribute__((ext_vector_type(8)))  _Float16 v8h;
typedef __attribute__((ext_vector_type(8)))  float    v8f;

static __device__ __forceinline__ v8f wmma_f16(v16h a, v16h b, v8f c) {
  // (neg_a, A, neg_b, B, c_mod, C, reuse_a, reuse_b) -> v_wmma_f32_16x16x32_f16
  return __builtin_amdgcn_wmma_f32_16x16x32_f16(false, a, false, b, (short)0, c, false, false);
}

// A fragment (16x32, MxK) from row-major [rows, ld] f16.
// Lane L (L<16): elems 0..7 = K 0..7, elems 8..15 = K 16..23 (row M=L).
// Lane L (>=16): elems 0..7 = K 8..15, elems 8..15 = K 24..31 (row M=L-16).
static __device__ __forceinline__ v16h load_frag_a(const _Float16* p, int ld,
                                                   int m0, int k0, int lane) {
  int m  = m0 + (lane & 15);
  int kh = (lane >> 4) * 8;
  const _Float16* row = p + (size_t)m * ld + k0;
  v8h lo = *(const v8h*)(row + kh);        // K = kh .. kh+7
  v8h hi = *(const v8h*)(row + 16 + kh);   // K = 16+kh .. 16+kh+7
  return __builtin_shufflevector(lo, hi, 0,1,2,3,4,5,6,7,8,9,10,11,12,13,14,15);
}

// B fragment (32x16, KxN) where matrix is stored (N,K) row-major (K contiguous).
// Lane L: N = L%16, elems 0..7 = K (L/16)*8 + e, elems 8..15 = K 16 + (L/16)*8 + (e-8).
static __device__ __forceinline__ v16h load_frag_b(const _Float16* p, int ld,
                                                   int n0, int k0, int lane) {
  int n  = n0 + (lane & 15);
  int kb = (lane >> 4) * 8;
  const _Float16* src = p + (size_t)n * ld + k0;
  v8h lo = *(const v8h*)(src + kb);
  v8h hi = *(const v8h*)(src + 16 + kb);
  return __builtin_shufflevector(lo, hi, 0,1,2,3,4,5,6,7,8,9,10,11,12,13,14,15);
}

// ---------------- 1. x -> f16, zero absmax ----------------
__global__ void k_prep(const float* __restrict__ x, _Float16* __restrict__ xh,
                       int n, int* absmax) {
  int i = blockIdx.x * blockDim.x + threadIdx.x;
  if (i == 0) *absmax = 0;
  if (i < n) xh[i] = (_Float16)x[i];
}

// ---------------- 2. per-row int8 quant-dequant of weights -> f16 ----------------
__global__ void k_quantw(const float* __restrict__ Wq, const float* __restrict__ Wk,
                         const float* __restrict__ Wv, const float* __restrict__ Wo,
                         _Float16* oq, _Float16* ok, _Float16* ov, _Float16* oo) {
  int wid  = blockIdx.x * (blockDim.x >> 5) + (threadIdx.x >> 5);
  int lane = threadIdx.x & 31;
  int mat  = wid >> 9;      // /512 rows
  int row  = wid & 511;
  const float* W = (mat == 0) ? Wq : (mat == 1) ? Wk : (mat == 2) ? Wv : Wo;
  _Float16*    O = (mat == 0) ? oq : (mat == 1) ? ok : (mat == 2) ? ov : oo;
  const float* r = W + (size_t)row * D_MODEL;
  float m = 0.f;
#pragma unroll
  for (int i = lane; i < D_MODEL; i += 32) m = fmaxf(m, fabsf(r[i]));
#pragma unroll
  for (int off = 16; off; off >>= 1) m = fmaxf(m, __shfl_xor(m, off, 32));
  float s   = fmaxf(m * (1.0f / 127.0f), 1e-8f);
  float inv = 1.0f / s;
  _Float16* o = O + (size_t)row * D_MODEL;
#pragma unroll
  for (int i = lane; i < D_MODEL; i += 32) {
    float q = rintf(r[i] * inv);
    q = fminf(fmaxf(q, -127.f), 127.f);
    o[i] = (_Float16)(q * s);
  }
}

// ---------------- 3. fused QKV projection (wave-tile WMMA GEMM) ----------------
// Q, K stored (B*H, T, dh);  V stored TRANSPOSED (B*H, dh, T) so that the
// P.V B-operand in attention is contiguous in K (two 16B chunks per lane).
__global__ __launch_bounds__(256) void k_qkv(
    const _Float16* __restrict__ xh,
    const _Float16* __restrict__ wq, const _Float16* __restrict__ wk,
    const _Float16* __restrict__ wv,
    const float* __restrict__ bq, const float* __restrict__ bk,
    const float* __restrict__ bv,
    _Float16* __restrict__ qb, _Float16* __restrict__ kb, _Float16* __restrict__ vbT) {
  int wid  = blockIdx.x * 8 + (threadIdx.x >> 5);
  int lane = threadIdx.x & 31;
  int sel  = wid / (512 * 32);          // 0=Q 1=K 2=V
  int rem  = wid % (512 * 32);
  int mt   = rem >> 5;
  int nt   = rem & 31;
  const _Float16* w    = (sel == 0) ? wq : (sel == 1) ? wk : wv;
  const float*    bias = (sel == 0) ? bq : (sel == 1) ? bk : bv;

  int m0 = mt * 16, n0 = nt * 16;
  v8f acc = {};
#pragma unroll
  for (int kt = 0; kt < 16; ++kt) {
    v16h a = load_frag_a(xh, D_MODEL, m0, kt * 32, lane);
    v16h b = load_frag_b(w,  D_MODEL, n0, kt * 32, lane);
    acc = wmma_f16(a, b, acc);
  }
  int   n  = n0 + (lane & 15);
  float bs = bias[n];
  int   h  = n >> 6, d = n & 63;
  int   hl = lane >> 4;
  if (sel == 2) {
#pragma unroll
    for (int r = 0; r < 8; ++r) {
      int m  = m0 + r + 8 * hl;
      int b_ = m >> 12, t = m & 4095;
      vbT[((size_t)(b_ * NUM_HEADS + h) * D_HEAD + d) * T_SEQ + t] = (_Float16)(acc[r] + bs);
    }
  } else {
    _Float16* dst = (sel == 0) ? qb : kb;
#pragma unroll
    for (int r = 0; r < 8; ++r) {
      int m  = m0 + r + 8 * hl;
      int b_ = m >> 12, t = m & 4095;
      dst[((size_t)(b_ * NUM_HEADS + h) * T_SEQ + t) * D_HEAD + d] = (_Float16)(acc[r] + bs);
    }
  }
}

// ---------------- 4. flash attention (one wave = one 16-query tile) ----------------
__global__ __launch_bounds__(256) void k_attn(
    const _Float16* __restrict__ q, const _Float16* __restrict__ k,
    const _Float16* __restrict__ vT, const int* __restrict__ mask,
    _Float16* __restrict__ ctx) {
  __shared__ _Float16 plds[8][16][32];   // per-wave P tile
  int wave = threadIdx.x >> 5;
  int lane = threadIdx.x & 31;
  int wid  = blockIdx.x * 8 + wave;
  int bh   = wid >> 8;                   // (b,h)
  int qt   = wid & 255;
  int b_   = bh >> 3;
  const _Float16* qbase  = q  + (size_t)bh * T_SEQ * D_HEAD;
  const _Float16* kbase  = k  + (size_t)bh * T_SEQ * D_HEAD;
  const _Float16* vTbase = vT + (size_t)bh * D_HEAD * T_SEQ;  // (dh, T)
  const int* mrow = mask + b_ * T_SEQ;

  v16h qf0 = load_frag_a(qbase, D_HEAD, qt * 16, 0,  lane);
  v16h qf1 = load_frag_a(qbase, D_HEAD, qt * 16, 32, lane);

  v8f acc0 = {}, acc1 = {}, acc2 = {}, acc3 = {};
  float rmax[8], rsum[8];
#pragma unroll
  for (int r = 0; r < 8; ++r) { rmax[r] = -1e30f; rsum[r] = 0.f; }

  int col = lane & 15;
  int hl  = lane >> 4;

  for (int s0 = 0; s0 < T_SEQ; s0 += 32) {
    // prefetch next key block (K rows and V^T d-rows 0..15)
    int sn = (s0 + 32) & (T_SEQ - 1);
    __builtin_prefetch(kbase + (size_t)sn * D_HEAD, 0, 1);
    __builtin_prefetch(vTbase + (size_t)col * T_SEQ + sn, 0, 1);

    // S = Q * K^T, two 16x16 column tiles
    v16h kb00 = load_frag_b(kbase, D_HEAD, s0,      0,  lane);
    v16h kb01 = load_frag_b(kbase, D_HEAD, s0,      32, lane);
    v16h kb10 = load_frag_b(kbase, D_HEAD, s0 + 16, 0,  lane);
    v16h kb11 = load_frag_b(kbase, D_HEAD, s0 + 16, 32, lane);
    v8f sa = {}, sb = {};
    sa = wmma_f16(qf0, kb00, sa); sa = wmma_f16(qf1, kb01, sa);
    sb = wmma_f16(qf0, kb10, sb); sb = wmma_f16(qf1, kb11, sb);

    int keep0 = mrow[s0 + col];
    int keep1 = mrow[s0 + 16 + col];

#pragma unroll
    for (int r = 0; r < 8; ++r) {
      float a0 = keep0 ? sa[r] * 0.125f : -1e30f;   // 1/sqrt(64)
      float a1 = keep1 ? sb[r] * 0.125f : -1e30f;
      float mx = fmaxf(a0, a1);
      mx = fmaxf(mx, __shfl_xor(mx, 1, 32));
      mx = fmaxf(mx, __shfl_xor(mx, 2, 32));
      mx = fmaxf(mx, __shfl_xor(mx, 4, 32));
      mx = fmaxf(mx, __shfl_xor(mx, 8, 32));
      float nm    = fmaxf(rmax[r], mx);
      float alpha = __expf(rmax[r] - nm);
      float p0 = keep0 ? __expf(a0 - nm) : 0.f;
      float p1 = keep1 ? __expf(a1 - nm) : 0.f;
      float ps = p0 + p1;
      ps += __shfl_xor(ps, 1, 32);
      ps += __shfl_xor(ps, 2, 32);
      ps += __shfl_xor(ps, 4, 32);
      ps += __shfl_xor(ps, 8, 32);
      rsum[r] = rsum[r] * alpha + ps;
      rmax[r] = nm;
      acc0[r] *= alpha; acc1[r] *= alpha; acc2[r] *= alpha; acc3[r] *= alpha;
      int m = r + 8 * hl;
      plds[wave][m][col]      = (_Float16)p0;
      plds[wave][m][16 + col] = (_Float16)p1;
    }
    asm volatile("s_wait_dscnt 0x0" ::: "memory");

    // P (16x32) from LDS; V^T rows are (N=d, K=key) row-major -> contiguous K
    v16h pf  = load_frag_a(&plds[wave][0][0], 32, 0, 0, lane);
    v16h vb0 = load_frag_b(vTbase, T_SEQ, 0,  s0, lane);
    v16h vb1 = load_frag_b(vTbase, T_SEQ, 16, s0, lane);
    v16h vb2 = load_frag_b(vTbase, T_SEQ, 32, s0, lane);
    v16h vb3 = load_frag_b(vTbase, T_SEQ, 48, s0, lane);
    acc0 = wmma_f16(pf, vb0, acc0);
    acc1 = wmma_f16(pf, vb1, acc1);
    acc2 = wmma_f16(pf, vb2, acc2);
    acc3 = wmma_f16(pf, vb3, acc3);
  }

  int hcol = (bh & 7) * D_HEAD;
#pragma unroll
  for (int r = 0; r < 8; ++r) {
    float inv = 1.0f / fmaxf(rsum[r], 1e-20f);
    int   m   = qt * 16 + r + 8 * hl;
    size_t o  = ((size_t)b_ * T_SEQ + m) * D_MODEL + hcol + col;
    ctx[o +  0] = (_Float16)(acc0[r] * inv);
    ctx[o + 16] = (_Float16)(acc1[r] * inv);
    ctx[o + 32] = (_Float16)(acc2[r] * inv);
    ctx[o + 48] = (_Float16)(acc3[r] * inv);
  }
}

// ---------------- 5. output projection + absmax ----------------
__global__ __launch_bounds__(256) void k_oproj(
    const _Float16* __restrict__ ctx, const _Float16* __restrict__ wo,
    const float* __restrict__ bo, float* __restrict__ out, int* absmax) {
  int wid  = blockIdx.x * 8 + (threadIdx.x >> 5);
  int lane = threadIdx.x & 31;
  int mt   = wid >> 5;
  int nt   = wid & 31;
  v8f acc = {};
#pragma unroll
  for (int kt = 0; kt < 16; ++kt) {
    v16h a = load_frag_a(ctx, D_MODEL, mt * 16, kt * 32, lane);
    v16h b = load_frag_b(wo,  D_MODEL, nt * 16, kt * 32, lane);
    acc = wmma_f16(a, b, acc);
  }
  int   n    = nt * 16 + (lane & 15);
  float bias = bo[n];
  float lmax = 0.f;
#pragma unroll
  for (int r = 0; r < 8; ++r) {
    int   m   = mt * 16 + r + 8 * (lane >> 4);
    float val = acc[r] + bias;
    out[(size_t)m * D_MODEL + n] = val;
    lmax = fmaxf(lmax, fabsf(val));
  }
#pragma unroll
  for (int off = 16; off; off >>= 1) lmax = fmaxf(lmax, __shfl_xor(lmax, off, 32));
  if (lane == 0) atomicMax(absmax, __float_as_int(lmax));  // positive floats order as ints
}

// ---------------- 6. per-tensor act quant-dequant ----------------
__global__ void k_quant_out(const float* __restrict__ in, const int* __restrict__ absmax,
                            float* __restrict__ out, int n) {
  int i = blockIdx.x * blockDim.x + threadIdx.x;
  if (i >= n) return;
  float s = fmaxf(__int_as_float(*absmax) * (1.0f / 127.0f), 1e-8f);
  float q = rintf(in[i] / s);
  q = fminf(fmaxf(q, -127.f), 127.f);
  out[i] = q * s;
}

extern "C" void kernel_launch(void* const* d_in, const int* in_sizes, int n_in,
                              void* d_out, int out_size, void* d_ws, size_t ws_size,
                              hipStream_t stream) {
  const float* x    = (const float*)d_in[0];
  const int*   mask = (const int*)  d_in[1];
  const float* Wq   = (const float*)d_in[2];
  const float* bq   = (const float*)d_in[3];
  const float* Wk   = (const float*)d_in[4];
  const float* bk   = (const float*)d_in[5];
  const float* Wv   = (const float*)d_in[6];
  const float* bv   = (const float*)d_in[7];
  const float* Wo   = (const float*)d_in[8];
  const float* bo   = (const float*)d_in[9];
  float* out = (float*)d_out;

  char* ws = (char*)d_ws;
  const size_t XH   = 0;                                   // 8192*512 f16
  const size_t WQH  = XH  + (size_t)MROWS * D_MODEL * 2;   // 4x 512*512 f16
  const size_t WKH  = WQH + (size_t)D_MODEL * D_MODEL * 2;
  const size_t WVH  = WKH + (size_t)D_MODEL * D_MODEL * 2;
  const size_t WOH  = WVH + (size_t)D_MODEL * D_MODEL * 2;
  const size_t QB   = WOH + (size_t)D_MODEL * D_MODEL * 2; // (16,4096,64) f16
  const size_t KB   = QB  + (size_t)MROWS * D_MODEL * 2;   // (16,4096,64) f16
  const size_t VBT  = KB  + (size_t)MROWS * D_MODEL * 2;   // (16,64,4096) f16 (transposed)
  const size_t CTX  = VBT + (size_t)MROWS * D_MODEL * 2;   // (B,T,512) f16
  const size_t O32  = CTX + (size_t)MROWS * D_MODEL * 2;   // (B,T,512) f32
  const size_t AMAX = O32 + (size_t)MROWS * D_MODEL * 4;

  _Float16* xh  = (_Float16*)(ws + XH);
  _Float16* wqh = (_Float16*)(ws + WQH);
  _Float16* wkh = (_Float16*)(ws + WKH);
  _Float16* wvh = (_Float16*)(ws + WVH);
  _Float16* woh = (_Float16*)(ws + WOH);
  _Float16* qb  = (_Float16*)(ws + QB);
  _Float16* kb  = (_Float16*)(ws + KB);
  _Float16* vbT = (_Float16*)(ws + VBT);
  _Float16* ctx = (_Float16*)(ws + CTX);
  float*    o32 = (float*)   (ws + O32);
  int*      am  = (int*)     (ws + AMAX);

  const int nelem = MROWS * D_MODEL;                 // 4194304

  k_prep<<<(nelem + 255) / 256, 256, 0, stream>>>(x, xh, nelem, am);
  k_quantw<<<(4 * 512) / 8, 256, 0, stream>>>(Wq, Wk, Wv, Wo, wqh, wkh, wvh, woh);
  k_qkv<<<(3 * 512 * 32) / 8, 256, 0, stream>>>(xh, wqh, wkh, wvh, bq, bk, bv, qb, kb, vbT);
  k_attn<<<(B_BATCH * NUM_HEADS * (T_SEQ / 16)) / 8, 256, 0, stream>>>(qb, kb, vbT, mask, ctx);
  k_oproj<<<(512 * 32) / 8, 256, 0, stream>>>(ctx, woh, bo, o32, am);
  k_quant_out<<<(nelem + 255) / 256, 256, 0, stream>>>(o32, am, out, nelem);
  (void)in_sizes; (void)n_in; (void)out_size; (void)ws_size;
}
